// AttnModel_1786706395630
// MI455X (gfx1250) — matrix operand order repaired and checked by reading, not compiled, not run on hardware
//
#include <hip/hip_runtime.h>

// ---------------------------------------------------------------------------
// Fused attention block for MI455X (gfx1250, wave32, WMMA).
//
// Roofline: after folding the K/V projections through the attention
// (N_src == 1), total compute is ~3 GFLOP and the kernel is bound by one
// streaming read of `seq` (268 MB) -> ~11.5 us at 23.3 TB/s.  Everything is
// fused into one kernel; no workspace round trips.  The seq-touching matmuls
// (scores = qk @ seq^T, ctx = attn @ seq) run on v_wmma_f32_16x16x32_bf16
// with f32 accumulation; softmax / LayerNorm / MLP stay f32 VALU.
// One workgroup (256 thr = 8 waves) processes 8 batches so the ~1.8 MB of
// weights are pulled from L2 once per 8 batches.
//
// Round-2 fixes: seq is converted to bf16 exactly once per element (staged
// both row-major `sSeq` and transposed `sSeqT` in LDS), so both WMMA k-loops
// are pure ds_load_b128 + v_wmma with no conversion VALU and no global
// re-reads of seq.
// ---------------------------------------------------------------------------

constexpr int FEAT   = 256;   // d_model == feat_dim
constexpr int NNGH   = 128;   // neighbors
constexpr int NH     = 4;     // heads
constexpr int DK     = 64;    // head dim
constexpr int GB     = 8;     // batches per workgroup
constexpr int TPB    = 256;   // threads per block (8 wave32 waves)
constexpr int TPITCH = 136;   // sSeqT row pitch (ushorts): 272B rows, 16B
                              // aligned, bank stride 68%64=4 -> conflict-free
constexpr int SPITCH = 264;   // sSeq row pitch (ushorts): 528B rows, 16B
                              // aligned, bank stride 132%64=4 -> conflict-free

typedef __attribute__((ext_vector_type(16))) __bf16 v16bf;
typedef __attribute__((ext_vector_type(8)))  float  v8f;

union BFrag {
  v16bf          v;
  unsigned short u[16];
  uint4          q[2];
};

__device__ __forceinline__ unsigned short f2bf(float f) {
  union { float f; unsigned u; } c; c.f = f;
  unsigned u = c.u;
  u += 0x7FFFu + ((u >> 16) & 1u);   // round-to-nearest-even
  return (unsigned short)(u >> 16);
}

// Dynamic LDS layout:
//   sSrc [GB][256] f32    sQf [GB][256] f32      sSc [4][128] f32
//   sCtx [GB][4][256] f32 sAO/sX/sXN/sH [GB][256] f32   sRed [16] f32
//   sQKb  [GB][4][256] bf16   sSeqT [256][TPITCH] bf16
//   sAttnB [4][128] bf16      sZero [16] bf16     sSeq [128][SPITCH] bf16
constexpr int SMEM_F32 = GB*256*6 + 4*128 + GB*4*256 + 16;             // floats
constexpr int SMEM_U16 = GB*4*256 + 256*TPITCH + 4*128 + 16 + NNGH*SPITCH;
constexpr int SMEM_BYTES = SMEM_F32*4 + SMEM_U16*2;                    // 238688

__global__ void __launch_bounds__(TPB, 1)
attn_fused_wmma(const float* __restrict__ src,  const float* __restrict__ seq,
                const int*   __restrict__ mask,
                const float* __restrict__ w_qs, const float* __restrict__ w_ks,
                const float* __restrict__ w_vs, const float* __restrict__ fc_w,
                const float* __restrict__ fc_b, const float* __restrict__ ln_g,
                const float* __restrict__ ln_b, const float* __restrict__ fc1_w,
                const float* __restrict__ fc1_b,const float* __restrict__ fc2_w,
                const float* __restrict__ fc2_b,
                float* __restrict__ out_z, float* __restrict__ out_attn,
                int Btot) {
  extern __shared__ char smem[];
  float* sSrc = (float*)smem;                       // [GB][256]
  float* sQf  = sSrc + GB*256;                      // [GB][256]
  float* sSc  = sQf  + GB*256;                      // [4][128]
  float* sCtx = sSc  + 4*128;                       // [GB][4][256]
  float* sAO  = sCtx + GB*4*256;                    // [GB][256]
  float* sX   = sAO  + GB*256;                      // [GB][256]
  float* sXN  = sX   + GB*256;                      // [GB][256]
  float* sH   = sXN  + GB*256;                      // [GB][256]
  float* sRed = sH   + GB*256;                      // [16] (mu,rstd per g)
  unsigned short* sQKb   = (unsigned short*)(sRed + 16);   // [GB][4][256]
  unsigned short* sSeqT  = sQKb  + GB*4*256;               // [256][TPITCH]
  unsigned short* sAttnB = sSeqT + 256*TPITCH;             // [4][128]
  unsigned short* sZero  = sAttnB + 4*128;                 // [16], stays zero
  unsigned short* sSeq   = sZero  + 16;                    // [128][SPITCH]

  const int t    = threadIdx.x;
  const int lane = t & 31;
  const int wv   = t >> 5;
  const int col  = lane & 15;     // WMMA B/D column within a fragment
  const int kh   = lane >> 4;     // half-wave selector in ISA layouts
  const int b0   = blockIdx.x * GB;
  const bool colv = (col < NH);

  // ---- Phase 1: stage src rows; zero the WMMA zero-block -----------------
  if (t < 16) sZero[t] = 0;
  #pragma unroll
  for (int g = 0; g < GB; ++g)
    sSrc[g*256 + t] = src[(size_t)(b0 + g)*256 + t];
  __syncthreads();

  // ---- Phase 2: q = src @ w_qs^T  (weight row stationary, 8 batches) -----
  {
    float acc[GB];
    #pragma unroll
    for (int g = 0; g < GB; ++g) acc[g] = 0.0f;
    const float* wr = w_qs + (size_t)t * 256;
    for (int i = 0; i < 256; i += 4) {
      const float4 w = *(const float4*)(wr + i);
      #pragma unroll
      for (int g = 0; g < GB; ++g) {
        const float4 s = *(const float4*)(sSrc + g*256 + i);
        acc[g] += w.x*s.x + w.y*s.y + w.z*s.z + w.w*s.w;
      }
    }
    #pragma unroll
    for (int g = 0; g < GB; ++g) sQf[g*256 + t] = acc[g];
  }
  __syncthreads();

  // ---- Phase 3: qk[h,i] = sum_d q[h,d] * w_ks[h*64+d, i]  (bf16 out) -----
  for (int h = 0; h < NH; ++h) {
    float acc[GB];
    #pragma unroll
    for (int g = 0; g < GB; ++g) acc[g] = 0.0f;
    for (int d = 0; d < DK; ++d) {
      const float w = w_ks[(size_t)(h*DK + d)*256 + t];   // coalesced in t
      #pragma unroll
      for (int g = 0; g < GB; ++g) acc[g] += w * sQf[g*256 + h*DK + d];
    }
    #pragma unroll
    for (int g = 0; g < GB; ++g) sQKb[(g*NH + h)*256 + t] = f2bf(acc[g]);
  }
  __syncthreads();

  // ---- Phase 4: per-batch attention (WMMA) -------------------------------
  for (int g = 0; g < GB; ++g) {
    const int b = b0 + g;
    const float* seqb = seq + (size_t)b * NNGH * 256;

    // 4a: single pass over seq_b: b128 global reads, convert to bf16 once,
    //     store row-major (b64 packed) and transposed (4x b16).
    {
      const int nsub = t >> 6;          // 0..3
      const int c4   = (t & 63) * 4;    // i-column group
      #pragma unroll 2
      for (int j = 0; j < 32; ++j) {
        const int n = j*4 + nsub;
        const float4 v4 = *(const float4*)(seqb + (size_t)n*256 + c4);
        union { unsigned short s[4]; uint2 u; } pk;
        pk.s[0] = f2bf(v4.x); pk.s[1] = f2bf(v4.y);
        pk.s[2] = f2bf(v4.z); pk.s[3] = f2bf(v4.w);
        *(uint2*)(sSeq + n*SPITCH + c4) = pk.u;     // row-major
        sSeqT[(c4+0)*TPITCH + n] = pk.s[0];          // transposed
        sSeqT[(c4+1)*TPITCH + n] = pk.s[1];
        sSeqT[(c4+2)*TPITCH + n] = pk.s[2];
        sSeqT[(c4+3)*TPITCH + n] = pk.s[3];
      }
    }
    __syncthreads();

    // 4b: scores^T[n][h] = seq_b x qk^T   (M=128 over 8 waves, N=16, K=256)
    {
      const int mt   = wv;                  // n-tile handled by this wave
      const int arow = mt*16 + col;         // neighbor index for A row
      const unsigned short* ap = sSeq + arow * SPITCH;
      // Uniform B addressing: invalid columns read the zero block, stride 0.
      const unsigned short* bp = colv ? (sQKb + (g*NH + col)*256 + kh*16)
                                      : sZero;
      const int bstep = colv ? 32 : 0;
      v8f acc = {0.f,0.f,0.f,0.f,0.f,0.f,0.f,0.f};
      #pragma unroll 4
      for (int k0 = 0; k0 < 256; k0 += 32) {
        BFrag A;
        A.q[0] = *(const uint4*)(ap + k0 + kh*16);
        A.q[1] = *(const uint4*)(ap + k0 + kh*16 + 8);
        BFrag Bf;
        Bf.q[0] = ((const uint4*)bp)[0];
        Bf.q[1] = ((const uint4*)bp)[1];
        bp += bstep;
        acc = __builtin_amdgcn_wmma_f32_16x16x32_bf16(
                  false, A.v, false, Bf.v, (short)0, acc, false, false);
      }
      if (colv) {
        const int mrow  = (4*b + col) & (Btot - 1);      // mask tiling (B pow2)
        const int nbase = mt*16 + kh*8;                  // D row = r + 8*kh
        const int* mp   = mask + (size_t)mrow * NNGH + nbase;
        const int4 m0 = *(const int4*)(mp);
        const int4 m1 = *(const int4*)(mp + 4);
        float* sc = sSc + col*NNGH + nbase;
        sc[0] = m0.x ? -1e10f : acc[0] * 0.125f;
        sc[1] = m0.y ? -1e10f : acc[1] * 0.125f;
        sc[2] = m0.z ? -1e10f : acc[2] * 0.125f;
        sc[3] = m0.w ? -1e10f : acc[3] * 0.125f;
        sc[4] = m1.x ? -1e10f : acc[4] * 0.125f;
        sc[5] = m1.y ? -1e10f : acc[5] * 0.125f;
        sc[6] = m1.z ? -1e10f : acc[6] * 0.125f;
        sc[7] = m1.w ? -1e10f : acc[7] * 0.125f;
      }
    }
    __syncthreads();

    // 4c: softmax over n (wave h handles head h), emit attn f32 + bf16
    if (wv < NH) {
      const int h = wv;
      float v0 = sSc[h*NNGH + lane];
      float v1 = sSc[h*NNGH + lane + 32];
      float v2 = sSc[h*NNGH + lane + 64];
      float v3 = sSc[h*NNGH + lane + 96];
      float m = fmaxf(fmaxf(v0, v1), fmaxf(v2, v3));
      #pragma unroll
      for (int s = 16; s > 0; s >>= 1) m = fmaxf(m, __shfl_xor(m, s, 32));
      float e0 = __expf(v0 - m), e1 = __expf(v1 - m);
      float e2 = __expf(v2 - m), e3 = __expf(v3 - m);
      float sum = e0 + e1 + e2 + e3;
      #pragma unroll
      for (int s = 16; s > 0; s >>= 1) sum += __shfl_xor(sum, s, 32);
      const float inv = 1.0f / sum;
      const float p0 = e0*inv, p1 = e1*inv, p2 = e2*inv, p3 = e3*inv;
      float* ao = out_attn + ((size_t)b*NH + h) * NNGH;
      ao[lane]      = p0;  ao[lane + 32] = p1;
      ao[lane + 64] = p2;  ao[lane + 96] = p3;
      sAttnB[h*NNGH + lane]      = f2bf(p0);
      sAttnB[h*NNGH + lane + 32] = f2bf(p1);
      sAttnB[h*NNGH + lane + 64] = f2bf(p2);
      sAttnB[h*NNGH + lane + 96] = f2bf(p3);
    }
    __syncthreads();

    // 4d: ctx^T[i][h] = seq^T x attn^T   (M=256 over 8 waves x2, N=16, K=128)
    #pragma unroll
    for (int half = 0; half < 2; ++half) {
      const int mt   = wv + half*8;          // i-tile
      const int irow = mt*16 + col;
      const unsigned short* arow = sSeqT + irow * TPITCH;
      const unsigned short* bp = colv ? (sAttnB + col*NNGH + kh*16) : sZero;
      const int bstep = colv ? 32 : 0;
      v8f acc = {0.f,0.f,0.f,0.f,0.f,0.f,0.f,0.f};
      #pragma unroll
      for (int k0 = 0; k0 < NNGH; k0 += 32) {
        BFrag A;
        A.q[0] = *(const uint4*)(arow + k0 + kh*16);
        A.q[1] = *(const uint4*)(arow + k0 + kh*16 + 8);
        BFrag Bf;
        Bf.q[0] = ((const uint4*)bp)[0];
        Bf.q[1] = ((const uint4*)bp)[1];
        bp += bstep;
        acc = __builtin_amdgcn_wmma_f32_16x16x32_bf16(
                  false, A.v, false, Bf.v, (short)0, acc, false, false);
      }
      if (colv) {
        const int ibase = mt*16 + kh*8;
        #pragma unroll
        for (int r = 0; r < 8; ++r)
          sCtx[(g*NH + col)*256 + ibase + r] = acc[r];
      }
    }
    __syncthreads();
  }

  // ---- Phase 5: out[o] = w_vs[o,:] . ctx[h(o),:] -------------------------
  {
    const int h = t >> 6;
    const float* wr = w_vs + (size_t)t * 256;
    float acc[GB];
    #pragma unroll
    for (int g = 0; g < GB; ++g) acc[g] = 0.0f;
    for (int i = 0; i < 256; i += 4) {
      const float4 w = *(const float4*)(wr + i);
      #pragma unroll
      for (int g = 0; g < GB; ++g) {
        const float4 c = *(const float4*)(sCtx + (g*NH + h)*256 + i);
        acc[g] += w.x*c.x + w.y*c.y + w.z*c.z + w.w*c.w;
      }
    }
    #pragma unroll
    for (int g = 0; g < GB; ++g) sAO[g*256 + t] = acc[g];
  }
  __syncthreads();

  // ---- Phase 6: fc projection + residual ---------------------------------
  {
    const float* wr = fc_w + (size_t)t * 256;
    float acc[GB];
    const float bb = fc_b[t];
    #pragma unroll
    for (int g = 0; g < GB; ++g) acc[g] = bb;
    for (int j = 0; j < 256; j += 4) {
      const float4 w = *(const float4*)(wr + j);
      #pragma unroll
      for (int g = 0; g < GB; ++g) {
        const float4 a = *(const float4*)(sAO + g*256 + j);
        acc[g] += w.x*a.x + w.y*a.y + w.z*a.z + w.w*a.w;
      }
    }
    #pragma unroll
    for (int g = 0; g < GB; ++g) sX[g*256 + t] = acc[g] + sSrc[g*256 + t];
  }
  __syncthreads();

  // ---- Phase 7: LayerNorm (wave g reduces batch g) -----------------------
  {
    const int g = wv;
    float s = 0.0f;
    #pragma unroll
    for (int p = 0; p < 8; ++p) s += sX[g*256 + lane + 32*p];
    #pragma unroll
    for (int x = 16; x > 0; x >>= 1) s += __shfl_xor(s, x, 32);
    const float mu = s * (1.0f/256.0f);
    float v = 0.0f;
    #pragma unroll
    for (int p = 0; p < 8; ++p) {
      const float d = sX[g*256 + lane + 32*p] - mu;
      v += d * d;
    }
    #pragma unroll
    for (int x = 16; x > 0; x >>= 1) v += __shfl_xor(v, x, 32);
    const float rstd = rsqrtf(v * (1.0f/256.0f) + 1e-5f);
    if (lane == 0) { sRed[g*2] = mu; sRed[g*2 + 1] = rstd; }
  }
  __syncthreads();
  {
    const float gg = ln_g[t], bb = ln_b[t];
    #pragma unroll
    for (int g = 0; g < GB; ++g)
      sXN[g*256 + t] = (sX[g*256 + t] - sRed[g*2]) * sRed[g*2 + 1] * gg + bb;
  }
  __syncthreads();

  // ---- Phase 8: fc1 over concat([x, src]) + ReLU -------------------------
  {
    const float* wr = fc1_w + (size_t)t * 512;
    float acc[GB];
    const float bb = fc1_b[t];
    #pragma unroll
    for (int g = 0; g < GB; ++g) acc[g] = bb;
    for (int j = 0; j < 256; j += 4) {
      const float4 w = *(const float4*)(wr + j);
      #pragma unroll
      for (int g = 0; g < GB; ++g) {
        const float4 x = *(const float4*)(sXN + g*256 + j);
        acc[g] += w.x*x.x + w.y*x.y + w.z*x.z + w.w*x.w;
      }
    }
    for (int j = 0; j < 256; j += 4) {
      const float4 w = *(const float4*)(wr + 256 + j);
      #pragma unroll
      for (int g = 0; g < GB; ++g) {
        const float4 s = *(const float4*)(sSrc + g*256 + j);
        acc[g] += w.x*s.x + w.y*s.y + w.z*s.z + w.w*s.w;
      }
    }
    #pragma unroll
    for (int g = 0; g < GB; ++g) sH[g*256 + t] = fmaxf(acc[g], 0.0f);
  }
  __syncthreads();

  // ---- Phase 9: fc2 -> z -------------------------------------------------
  {
    const float* wr = fc2_w + (size_t)t * 256;
    float acc[GB];
    const float bb = fc2_b[t];
    #pragma unroll
    for (int g = 0; g < GB; ++g) acc[g] = bb;
    for (int j = 0; j < 256; j += 4) {
      const float4 w = *(const float4*)(wr + j);
      #pragma unroll
      for (int g = 0; g < GB; ++g) {
        const float4 h = *(const float4*)(sH + g*256 + j);
        acc[g] += w.x*h.x + w.y*h.y + w.z*h.z + w.w*h.w;
      }
    }
    #pragma unroll
    for (int g = 0; g < GB; ++g)
      out_z[(size_t)(b0 + g)*256 + t] = acc[g];
  }
}

extern "C" void kernel_launch(void* const* d_in, const int* in_sizes, int n_in,
                              void* d_out, int out_size, void* d_ws, size_t ws_size,
                              hipStream_t stream) {
  const float* src   = (const float*)d_in[0];
  const float* seq   = (const float*)d_in[1];
  const int*   mask  = (const int*)  d_in[2];
  const float* w_qs  = (const float*)d_in[3];
  const float* w_ks  = (const float*)d_in[4];
  const float* w_vs  = (const float*)d_in[5];
  const float* fc_w  = (const float*)d_in[6];
  const float* fc_b  = (const float*)d_in[7];
  const float* ln_g  = (const float*)d_in[8];
  const float* ln_b  = (const float*)d_in[9];
  const float* fc1_w = (const float*)d_in[10];
  const float* fc1_b = (const float*)d_in[11];
  const float* fc2_w = (const float*)d_in[12];
  const float* fc2_b = (const float*)d_in[13];

  const int Btot = in_sizes[0] / FEAT;          // 2048 (power of two assumed)
  float* out_z    = (float*)d_out;              // [B,1,256]
  float* out_attn = out_z + (size_t)Btot*FEAT;  // [B,1,4,128]

  (void)d_ws; (void)ws_size; (void)n_in; (void)out_size;

  static_assert(SMEM_BYTES <= 320*1024, "LDS budget");
  hipFuncSetAttribute((const void*)attn_fused_wmma,
                      hipFuncAttributeMaxDynamicSharedMemorySize, SMEM_BYTES);

  dim3 grid(Btot / GB), block(TPB);
  attn_fused_wmma<<<grid, block, SMEM_BYTES, stream>>>(
      src, seq, mask, w_qs, w_ks, w_vs, fc_w, fc_b, ln_g, ln_b,
      fc1_w, fc1_b, fc2_w, fc2_b, out_z, out_attn, Btot);
}
